// ScaledDotProductAttention_28278064677159
// MI455X (gfx1250) — matrix operand order repaired and checked by reading, not compiled, not run on hardware
//
#include <hip/hip_runtime.h>

#define SLEN 4096
#define DH   64
#define BATCH 8
#define WPB  8   // waves per block (all same batch; share K/V through WGP$)

typedef __attribute__((ext_vector_type(16))) __bf16 v16bf;
typedef __attribute__((ext_vector_type(8)))  __bf16 v8bf;
typedef __attribute__((ext_vector_type(8)))  float  v8f;

// Load 16 contiguous bf16 (32 bytes) into a v16bf WMMA operand.
__device__ __forceinline__ v16bf ld_bf16x16(const __bf16* p) {
    v8bf lo = *reinterpret_cast<const v8bf*>(p);
    v8bf hi = *reinterpret_cast<const v8bf*>(p + 8);
    return __builtin_shufflevector(lo, hi, 0,1,2,3,4,5,6,7,8,9,10,11,12,13,14,15);
}

// Build an A-fragment from two adjacent 8-element chunks (16B each).
__device__ __forceinline__ v16bf ld_a_frag(const __bf16* lo_p, const __bf16* hi_p) {
    v8bf lo = *reinterpret_cast<const v8bf*>(lo_p);
    v8bf hi = *reinterpret_cast<const v8bf*>(hi_p);
    return __builtin_shufflevector(lo, hi, 0,1,2,3,4,5,6,7,8,9,10,11,12,13,14,15);
}

// max with a DPP row-rotate (stays within the 16-lane row half).
template <int CTRL>
__device__ __forceinline__ float dpp_ror_max(float x) {
    int y = __builtin_amdgcn_update_dpp(0, __builtin_bit_cast(int, x),
                                        CTRL, 0xF, 0xF, true);
    return fmaxf(x, __builtin_bit_cast(float, y));
}
__device__ __forceinline__ float rowmax16(float x) {
    x = dpp_ror_max<0x121>(x);  // row_ror:1
    x = dpp_ror_max<0x122>(x);  // row_ror:2
    x = dpp_ror_max<0x124>(x);  // row_ror:4
    x = dpp_ror_max<0x128>(x);  // row_ror:8
    return x;
}

// ---------------------------------------------------------------------------
// Prep: Qb = bf16(Q*scale*log2e) [B,S,D]; Kb = bf16(K) [B,S,D];
//       Vt = bf16(V^T) [B,D,S]. Scores end up in the log2 domain -> exp2f.
// ---------------------------------------------------------------------------
__global__ void attn_prep_kernel(const float* __restrict__ Q,
                                 const float* __restrict__ K,
                                 const float* __restrict__ V,
                                 __bf16* __restrict__ Qb,
                                 __bf16* __restrict__ Kb,
                                 __bf16* __restrict__ Vt) {
    const float scale = 0.125f * 1.44269504088896340736f; // 1/sqrt(64) * log2(e)
    size_t idx = (size_t)blockIdx.x * blockDim.x + threadIdx.x;
    size_t N = (size_t)BATCH * SLEN * DH;
    if (idx >= N) return;
    Qb[idx] = (__bf16)(Q[idx] * scale);
    Kb[idx] = (__bf16)(K[idx]);
    size_t b = idx / ((size_t)SLEN * DH);
    size_t r = idx - b * ((size_t)SLEN * DH);
    size_t s = r / DH;
    size_t d = r - s * DH;
    Vt[b * ((size_t)DH * SLEN) + d * SLEN + s] = (__bf16)(V[idx]);
}

// Mask f32 -> bf16 (halves the dominant repeated-read traffic; 32MB in L2).
__global__ void mask_prep_kernel(const float* __restrict__ mask,
                                 __bf16* __restrict__ Mb) {
    size_t idx = (size_t)blockIdx.x * blockDim.x + threadIdx.x;
    if (idx >= (size_t)SLEN * SLEN) return;
    Mb[idx] = (__bf16)mask[idx];
}

// ---------------------------------------------------------------------------
// Flash attention, post-softmax multiplicative mask fused into the streaming
// exp2 weights. One wave = one 16-row Q tile, KV tile = 64 columns; the 8
// waves of a block walk the same K/V tiles in lockstep (barrier per tile) so
// K/V traffic is served out of the WGP$. Denominator via ones-column WMMA.
// Row max maintained lazily behind a wave-uniform ballot.
// ---------------------------------------------------------------------------
template <typename MT>
__global__ __launch_bounds__(WPB * 32)
void attn_main_kernel(const __bf16* __restrict__ Qb,
                      const __bf16* __restrict__ Kb,
                      const __bf16* __restrict__ Vt,
                      const MT*     __restrict__ mask,
                      float* __restrict__ out) {
    __shared__ __align__(16) __bf16 pbuf[WPB][16 * 64];

    const int lane = threadIdx.x & 31;
    const int w    = threadIdx.x >> 5;
    const int tile = blockIdx.x * WPB + w;          // 0 .. B*S/16-1
    const int b    = tile >> 8;                     // 256 q-tiles per batch
    const int qBase = (tile & 255) << 4;
    const int n = lane & 15;
    const int h = lane >> 4;

    const __bf16* Kbase = Kb + (size_t)b * SLEN * DH;
    const __bf16* Vbase = Vt + (size_t)b * DH * SLEN;
    const MT*     mbase = mask + (size_t)(qBase + 8 * h) * SLEN + n;

    // A-fragments of Q (16x32 each, K split 0..31 / 32..63)
    const __bf16* Qrow = Qb + ((size_t)b * SLEN + qBase + n) * DH;
    v16bf aq0 = ld_a_frag(Qrow + h * 16,      Qrow + 8 + h * 16);
    v16bf aq1 = ld_a_frag(Qrow + 32 + h * 16, Qrow + 40 + h * 16);

    // all-ones B fragment (denominator accumulator feed)
    v16bf ones;
    #pragma unroll
    for (int i = 0; i < 16; ++i) ones[i] = (__bf16)1.0f;

    v8f zero = {};
    v8f acc[5];                       // [0..3] = O chunks, [4] = denominator
    #pragma unroll
    for (int i = 0; i < 5; ++i) acc[i] = zero;
    float mrow[8];
    #pragma unroll
    for (int r = 0; r < 8; ++r) mrow[r] = -1e30f;

    for (int kv = 0; kv < SLEN; kv += 64) {
        __syncthreads();   // keep the 8 waves inside the WGP$ reuse window

        // ---- scores: S = Qb * Kb^T, 16x64 tile as four 16x16 fragments ----
        v8f c[4];
        #pragma unroll
        for (int j = 0; j < 4; ++j) {
            const __bf16* kp = Kbase + (size_t)(kv + j * 16 + n) * DH;
            v16bf bk0 = ld_bf16x16(kp + h * 16);
            v16bf bk1 = ld_bf16x16(kp + 32 + h * 16);
            v8f cj = zero;
            cj = __builtin_amdgcn_wmma_f32_16x16x32_bf16(false, aq0, false, bk0, (short)0, cj, false, false);
            cj = __builtin_amdgcn_wmma_f32_16x16x32_bf16(false, aq1, false, bk1, (short)0, cj, false, false);
            c[j] = cj;
        }

        // ---- lazy running-max check: cheap per-lane partial max + ballot ----
        float pm[8];
        bool anych = false;
        #pragma unroll
        for (int r = 0; r < 8; ++r) {
            pm[r] = fmaxf(fmaxf(c[0][r], c[1][r]), fmaxf(c[2][r], c[3][r]));
            anych = anych || (pm[r] > mrow[r]);
        }
        if (__ballot(anych) != 0ull) {   // wave-uniform; rare after warmup
            #pragma unroll
            for (int r = 0; r < 8; ++r) {
                float newm = fmaxf(mrow[r], rowmax16(pm[r]));
                float corr = exp2f(mrow[r] - newm);
                mrow[r] = newm;
                #pragma unroll
                for (int dc = 0; dc < 5; ++dc) acc[dc][r] *= corr;
            }
        }

        // ---- weights: P = exp2(score - mrow) * mask, stashed to LDS ----
        #pragma unroll
        for (int r = 0; r < 8; ++r) {
            const MT* mp = mbase + (size_t)r * SLEN + kv;
            #pragma unroll
            for (int j = 0; j < 4; ++j) {
                float e = exp2f(c[j][r] - mrow[r]) * (float)mp[j * 16];
                pbuf[w][(r + 8 * h) * 64 + j * 16 + n] = (__bf16)e;
            }
        }
        asm volatile("s_wait_dscnt 0" ::: "memory");

        // reload P as two A-fragments (row m = lane%16)
        const __bf16* pr = &pbuf[w][n * 64];
        v16bf ap0 = ld_a_frag(pr + h * 16,      pr + 8 + h * 16);
        v16bf ap1 = ld_a_frag(pr + 32 + h * 16, pr + 40 + h * 16);

        // ---- O += P * V via transposed V; denominator += P * ones ----
        #pragma unroll
        for (int dc = 0; dc < 4; ++dc) {
            const __bf16* vp = Vbase + (size_t)(dc * 16 + n) * SLEN + kv + h * 16;
            v16bf bv0 = ld_bf16x16(vp);
            v16bf bv1 = ld_bf16x16(vp + 32);
            acc[dc] = __builtin_amdgcn_wmma_f32_16x16x32_bf16(false, ap0, false, bv0, (short)0, acc[dc], false, false);
            acc[dc] = __builtin_amdgcn_wmma_f32_16x16x32_bf16(false, ap1, false, bv1, (short)0, acc[dc], false, false);
        }
        acc[4] = __builtin_amdgcn_wmma_f32_16x16x32_bf16(false, ap0, false, ones, (short)0, acc[4], false, false);
        acc[4] = __builtin_amdgcn_wmma_f32_16x16x32_bf16(false, ap1, false, ones, (short)0, acc[4], false, false);
    }

    // ---- epilogue: divide by mask-weighted denominator, store f32 ----
    #pragma unroll
    for (int r = 0; r < 8; ++r) {
        float inv = 1.0f / acc[4][r];
        size_t row = ((size_t)b * SLEN + qBase + r + 8 * h) * DH;
        #pragma unroll
        for (int dc = 0; dc < 4; ++dc) {
            out[row + dc * 16 + n] = acc[dc][r] * inv;
        }
    }
}

extern "C" void kernel_launch(void* const* d_in, const int* in_sizes, int n_in,
                              void* d_out, int out_size, void* d_ws, size_t ws_size,
                              hipStream_t stream) {
    const float* Q    = (const float*)d_in[0];
    const float* K    = (const float*)d_in[1];
    const float* V    = (const float*)d_in[2];
    const float* mask = (const float*)d_in[3];
    float* out = (float*)d_out;

    const size_t N  = (size_t)BATCH * SLEN * DH;       // 2 Mi elements
    const size_t NM = (size_t)SLEN * SLEN;             // 16 Mi elements
    const size_t need_base = 3 * N * sizeof(__bf16);   // Qb + Kb + Vt  (12 MB)
    const size_t need_mask = need_base + NM * sizeof(__bf16); // + 32 MB

    char* ws = (char*)d_ws;
    __bf16* Qb = (__bf16*)(ws);
    __bf16* Kb = (__bf16*)(ws + N * sizeof(__bf16));
    __bf16* Vt = (__bf16*)(ws + 2 * N * sizeof(__bf16));
    __bf16* Mb = (__bf16*)(ws + need_base);

    // Prep: convert + transpose into workspace
    {
        int threads = 256;
        int blocks = (int)((N + threads - 1) / threads);
        attn_prep_kernel<<<blocks, threads, 0, stream>>>(Q, K, V, Qb, Kb, Vt);
    }

    const int tiles  = BATCH * (SLEN / 16);            // 2048
    const int blocks = tiles / WPB;                    // 256

    if (ws_size >= need_mask) {
        int threads = 256;
        int mblocks = (int)((NM + threads - 1) / threads);
        mask_prep_kernel<<<mblocks, threads, 0, stream>>>(mask, Mb);
        attn_main_kernel<__bf16><<<blocks, WPB * 32, 0, stream>>>(Qb, Kb, Vt, Mb, out);
    } else {
        attn_main_kernel<float><<<blocks, WPB * 32, 0, stream>>>(Qb, Kb, Vt, mask, out);
    }
}
